// LinearTriParser_9302899163650
// MI455X (gfx1250) — matrix operand order repaired and checked by reading, not compiled, not run on hardware
//
#include <hip/hip_runtime.h>
#include <hip/hip_bf16.h>

// Problem constants (match reference): B=2, S=128, H=1024, A=256, C=14
static constexpr int kB = 2;
static constexpr int kS = 128;
static constexpr int kH = 1024;
static constexpr int kA = 256;
static constexpr int kC = 14;
static constexpr int kM = kB * kS;   // 256 flattened rows

typedef __bf16 bf16_t;
typedef bf16_t v16bf __attribute__((ext_vector_type(16)));
typedef bf16_t v8bf  __attribute__((ext_vector_type(8)));
typedef float  v8f   __attribute__((ext_vector_type(8)));
typedef int    v4i   __attribute__((ext_vector_type(4)));

typedef __attribute__((address_space(1))) v4i* gv4i_ptr;  // global 16B chunk
typedef __attribute__((address_space(3))) v4i* lv4i_ptr;  // LDS 16B chunk

// CDNA5 async global->LDS copy path (guarded; falls back to vector copy)
#if defined(__has_builtin)
#if __has_builtin(__builtin_amdgcn_global_load_async_to_lds_b128) && \
    __has_builtin(__builtin_amdgcn_s_wait_asynccnt)
#define USE_ASYNC_LDS 1
#endif
#endif
#ifndef USE_ASYNC_LDS
#define USE_ASYNC_LDS 0
#endif

// ---------------------------------------------------------------------------
// fp32 -> bf16 conversion (memory activations, row-major preserved)
// ---------------------------------------------------------------------------
__global__ void f32_to_bf16_kernel(const float* __restrict__ in,
                                   bf16_t* __restrict__ out, int n) {
  int i = blockIdx.x * blockDim.x + threadIdx.x;
  if (i < n) out[i] = (bf16_t)in[i];
}

// ---------------------------------------------------------------------------
// Weight repack: W[K,N] f32 row-major -> Wcm[N,K] bf16 (column-major).
// Makes every B-fragment a 32B contiguous load per lane.
// ---------------------------------------------------------------------------
__global__ void repack_w_kernel(const float* __restrict__ W,
                                bf16_t* __restrict__ Wcm, int K, int N) {
  int idx = blockIdx.x * blockDim.x + threadIdx.x;  // n*K + k (exact grid)
  int n = idx / K;
  int k = idx % K;
  Wcm[idx] = (bf16_t)W[(size_t)k * N + n];
}

// ---------------------------------------------------------------------------
// WMMA GEMM: D = act(A[M,K](bf16,row-major) x Wcm[N,K](bf16,col-major) + bias)
// Block = 4 waves; block covers one 16-row M strip x 64 N columns.
// A panel (16 x K, contiguous in A) staged once into LDS (async copy),
// shared by all 4 waves; B read as two global_load_b128 per K-step.
// One v_wmma_f32_16x16x32_bf16 per 16x16x32 step.
// ---------------------------------------------------------------------------
template <bool RELU, bool OUTBF16>
__global__ __launch_bounds__(128) void wmma_gemm_kernel(
    const bf16_t* __restrict__ A, const bf16_t* __restrict__ Wcm,
    const float* __restrict__ bias, void* __restrict__ outp,
    int N, int K) {
  extern __shared__ char smem_raw[];
  bf16_t* sA = (bf16_t*)smem_raw;                    // [16][K] row-major

  const int tm = blockIdx.y;                          // M tile
  const int tn = blockIdx.x * 4 + threadIdx.y;        // N tile (per wave)
  const int lane = threadIdx.x;                       // 0..31
  const int half = lane >> 4;
  const int r    = lane & 15;
  const int tid  = threadIdx.y * 32 + threadIdx.x;    // 0..127

  // ---- Stage contiguous 16-row A panel into LDS (16*K bf16 = 32*K bytes) --
  {
    const bf16_t* gA = A + (size_t)tm * 16 * K;       // contiguous block
    const int nchunks = (16 * K) / 8;                 // 16B chunks
#if USE_ASYNC_LDS
    for (int c = tid; c < nchunks; c += 128) {
      __builtin_amdgcn_global_load_async_to_lds_b128(
          (gv4i_ptr)(gA + (size_t)c * 8),
          (lv4i_ptr)(sA + (size_t)c * 8),
          0, 0);
    }
    __builtin_amdgcn_s_wait_asynccnt(0);
#else
    for (int c = tid; c < nchunks; c += 128) {
      *(v8bf*)(sA + (size_t)c * 8) = *(const v8bf*)(gA + (size_t)c * 8);
    }
#endif
    __syncthreads();
  }

  const int bcol = tn * 16 + r;                       // output column

  v8f acc;
  {
    const float bv = bias[bcol];
#pragma unroll
    for (int v = 0; v < 8; ++v) acc[v] = bv;
  }

  const bf16_t* aRow = sA + (size_t)r * K;            // LDS row for this lane
  const bf16_t* bCol = Wcm + (size_t)bcol * K;        // bf16 column, contiguous

  for (int k0 = 0; k0 < K; k0 += 32) {
    // A fragment 16x32: element e -> K = k0 + half*16 + e
    v8bf a_lo = *(const v8bf*)(aRow + k0 + half * 16);
    v8bf a_hi = *(const v8bf*)(aRow + k0 + half * 16 + 8);
    v16bf af = __builtin_shufflevector(a_lo, a_hi, 0, 1, 2, 3, 4, 5, 6, 7,
                                       8, 9, 10, 11, 12, 13, 14, 15);
    // B fragment 32x16: element e -> K = k0 + half*16 + e  (32B contiguous)
    v8bf b_lo = *(const v8bf*)(bCol + k0 + half * 16);
    v8bf b_hi = *(const v8bf*)(bCol + k0 + half * 16 + 8);
    v16bf bfr = __builtin_shufflevector(b_lo, b_hi, 0, 1, 2, 3, 4, 5, 6, 7,
                                        8, 9, 10, 11, 12, 13, 14, 15);

    acc = __builtin_amdgcn_wmma_f32_16x16x32_bf16(
        /*neg_a=*/false, af, /*neg_b=*/false, bfr,
        /*c_mod=*/(short)0, acc, /*reuse_a=*/false, /*reuse_b=*/false);
  }

#pragma unroll
  for (int v = 0; v < 8; ++v) {
    const int row = tm * 16 + v + half * 8;
    float x = acc[v];
    if (RELU) x = fmaxf(x, 0.0f);
    if (OUTBF16)
      ((bf16_t*)outp)[(size_t)row * N + bcol] = (bf16_t)x;
    else
      ((float*)outp)[(size_t)row * N + bcol] = x;
  }
}

// ---------------------------------------------------------------------------
// Projection: out[row,c] = b[c] + sum_k Y[row,k] * W[k,c]  (Y:[256,256], W:[256,14])
// ---------------------------------------------------------------------------
__global__ void proj_kernel(const float* __restrict__ Y, const float* __restrict__ W,
                            const float* __restrict__ b, float* __restrict__ out) {
  const int idx = blockIdx.x * blockDim.x + threadIdx.x;  // exact grid 256*14
  const int row = idx / kC;
  const int c   = idx % kC;
  float s = b[c];
  const float* y = Y + row * kA;
#pragma unroll 8
  for (int k = 0; k < kA; ++k) s = fmaf(y[k], W[k * kC + c], s);
  out[idx] = s;
}

// ---------------------------------------------------------------------------
// Final: per (b,c) prefix-scan over k of e=exp(sm0), e*sm1, sm1; then emit
// out[b,i,j,c] = sh1[b,i,c] + st1[b,j,c] + uni[c] +
//                ( i<=j ? (Pes[j]-Pes[i-1])/(Pe[j]-Pe[i-1]) : mean_k sm1 )
// ---------------------------------------------------------------------------
__global__ __launch_bounds__(kS) void final_kernel(
    const float* __restrict__ sh1, const float* __restrict__ st1,
    const float* __restrict__ sm0, const float* __restrict__ sm1,
    const float* __restrict__ uni, float* __restrict__ out) {
  const int b = blockIdx.x / kC;
  const int c = blockIdx.x % kC;
  const int k = threadIdx.x;  // 0..127, doubles as output column j

  __shared__ float Pe[kS], Pes[kS], Psm[kS], sH[kS], sT[kS];

  const int base = (b * kS + k) * kC + c;
  const float e0   = expf(sm0[base]);
  const float sm1v = sm1[base];
  sH[k]  = sh1[base];
  sT[k]  = st1[base];
  Pe[k]  = e0;
  Pes[k] = e0 * sm1v;
  Psm[k] = sm1v;
  __syncthreads();

  for (int off = 1; off < kS; off <<= 1) {
    float a0 = 0.f, a1 = 0.f, a2 = 0.f;
    if (k >= off) { a0 = Pe[k - off]; a1 = Pes[k - off]; a2 = Psm[k - off]; }
    __syncthreads();
    Pe[k] += a0; Pes[k] += a1; Psm[k] += a2;
    __syncthreads();
  }

  const float meanSm1 = Psm[kS - 1] * (1.0f / kS);
  const float u = uni[c];
  const int j = k;
  const float pej = Pe[j], pesj = Pes[j];
  const float tj = sT[j] + u;

  for (int i = 0; i < kS; ++i) {
    float w;
    if (i <= j) {
      const float pe0 = (i > 0) ? Pe[i - 1] : 0.0f;
      const float ps0 = (i > 0) ? Pes[i - 1] : 0.0f;
      w = (pesj - ps0) / (pej - pe0);
    } else {
      w = meanSm1;
    }
    out[(((size_t)b * kS + i) * kS + j) * kC + c] = sH[i] + tj + w;
  }
}

// ---------------------------------------------------------------------------
// Launch
// ---------------------------------------------------------------------------
extern "C" void kernel_launch(void* const* d_in, const int* in_sizes, int n_in,
                              void* d_out, int out_size, void* d_ws, size_t ws_size,
                              hipStream_t stream) {
  (void)in_sizes; (void)n_in; (void)out_size; (void)ws_size;

  const float* memory = (const float*)d_in[0];
  const float* uni = (const float*)d_in[25];

  char* ws = (char*)d_ws;
  bf16_t* Xbf  = (bf16_t*)(ws);                     // 256*1024 bf16 = 512 KB
  bf16_t* W1cm = (bf16_t*)(ws + (512 << 10));       // 1024*256 bf16 = 512 KB
  bf16_t* W2cm = (bf16_t*)(ws + (1024 << 10));      // 256*256  bf16 = 128 KB
  bf16_t* Y1   = (bf16_t*)(ws + (1152 << 10));      // 256*256  bf16 = 128 KB
  float*  Y2   = (float*)(ws + (1280 << 10));       // 256*256  f32  = 256 KB
  float*  sh1  = (float*)(ws + (1536 << 10));       // 256*14 f32 each
  float*  st1  = sh1 + kM * kC;
  float*  sm0  = st1 + kM * kC;
  float*  sm1  = sm0 + kM * kC;

  // memory -> bf16
  f32_to_bf16_kernel<<<(kM * kH) / 256, 256, 0, stream>>>(memory, Xbf, kM * kH);

  const dim3 gblock(32, 4);
  const dim3 ggrid(kA / 64, kM / 16);               // (4, 16)
  const size_t sh1b = (size_t)16 * kH * sizeof(bf16_t);  // 32 KB (GEMM1)
  const size_t sh2b = (size_t)16 * kA * sizeof(bf16_t);  //  8 KB (GEMM2)

  for (int br = 0; br < 3; ++br) {
    const float* W1  = (const float*)d_in[1 + 8 * br + 0];
    const float* b1  = (const float*)d_in[1 + 8 * br + 1];
    const float* W2  = (const float*)d_in[1 + 8 * br + 2];
    const float* b2  = (const float*)d_in[1 + 8 * br + 3];
    const float* s1W = (const float*)d_in[1 + 8 * br + 6];
    const float* s1b = (const float*)d_in[1 + 8 * br + 7];

    // Repack weights to bf16 column-major
    repack_w_kernel<<<(kH * kA) / 256, 256, 0, stream>>>(W1, W1cm, kH, kA);
    repack_w_kernel<<<(kA * kA) / 256, 256, 0, stream>>>(W2, W2cm, kA, kA);

    // Y1 = relu(X @ W1 + b1)  [256,1024]x[1024,256] -> bf16
    wmma_gemm_kernel<true, true><<<ggrid, gblock, sh1b, stream>>>(
        Xbf, W1cm, b1, (void*)Y1, kA, kH);
    // Y2 = Y1 @ W2 + b2       [256,256]x[256,256]  -> f32
    wmma_gemm_kernel<false, false><<<ggrid, gblock, sh2b, stream>>>(
        Y1, W2cm, b2, (void*)Y2, kA, kA);

    if (br == 0) {
      proj_kernel<<<kC, kM, 0, stream>>>(Y2, s1W, s1b, sh1);
    } else if (br == 1) {
      proj_kernel<<<kC, kM, 0, stream>>>(Y2, s1W, s1b, st1);
    } else {
      const float* s0W = (const float*)d_in[1 + 8 * br + 4];
      const float* s0b = (const float*)d_in[1 + 8 * br + 5];
      proj_kernel<<<kC, kM, 0, stream>>>(Y2, s0W, s0b, sm0);
      proj_kernel<<<kC, kM, 0, stream>>>(Y2, s1W, s1b, sm1);
    }
  }

  final_kernel<<<kB * kC, kS, 0, stream>>>(sh1, st1, sm0, sm1, uni, (float*)d_out);
}
